// GCN_16003048145328
// MI455X (gfx1250) — matrix operand order repaired and checked by reading, not compiled, not run on hardware
//
#include <hip/hip_runtime.h>

#define NN 50000
#define NE 800000
#define INF 256
#define HID 64
#define OUTF 32

typedef __attribute__((ext_vector_type(16))) __bf16 v16bf;
typedef __attribute__((ext_vector_type(8)))  float  v8f;
typedef int vi4 __attribute__((vector_size(16)));

#define AS1 __attribute__((address_space(1)))
#define AS3 __attribute__((address_space(3)))

#if defined(__AMDGCN__) && __has_builtin(__builtin_amdgcn_global_load_async_to_lds_b128) && \
    __has_builtin(__builtin_amdgcn_s_wait_asynccnt)
#define USE_ASYNC_LDS 1
#else
#define USE_ASYNC_LDS 0
#endif

// ---------------- utility kernels ----------------

__global__ void zero_f32(float* __restrict__ p, long n) {
  long i = (long)blockIdx.x * blockDim.x + threadIdx.x;
  long s = (long)gridDim.x * blockDim.x;
  for (; i < n; i += s) p[i] = 0.0f;
}

__global__ void degree_kernel(const int* __restrict__ src, const int* __restrict__ dst,
                              float* __restrict__ deg_out, float* __restrict__ deg_in, int E) {
  int i = blockIdx.x * blockDim.x + threadIdx.x;
  if (i < E) {
    atomicAdd(&deg_out[src[i]], 1.0f);
    atomicAdd(&deg_in[dst[i]], 1.0f);
  }
}

__global__ void norm_kernel(float* __restrict__ deg, int n) {
  int i = blockIdx.x * blockDim.x + threadIdx.x;
  if (i < n) deg[i] = rsqrtf(fmaxf(deg[i], 1.0f));
}

// ---------------- fragment packing (fused f32->bf16 conversion) ----------------
// A [M x K] row-major  ->  Ap fragments: frag (mtile, kb) = 32 lanes x v16bf, contiguous.
// Wave32 16-bit A layout: lane = g*16 + m ; element e of the v16bf per (g, kk):
//   kk<8:(g0,e=kk)  8..15:(g1,e=kk-8)  16..23:(g0,e=kk-8)  24..31:(g1,e=kk-16)
__global__ void pack_A_bf16(const float* __restrict__ X, __bf16* __restrict__ Ap,
                            long total, int K) {
  long i = (long)blockIdx.x * blockDim.x + threadIdx.x;
  long s = (long)gridDim.x * blockDim.x;
  int KB = K >> 5;
  for (; i < total; i += s) {
    int row = (int)(i / K), k = (int)(i % K);
    int mtile = row >> 4, m = row & 15;
    int kb = k >> 5, kk = k & 31;
    int g, e;
    if (kk < 8)       { g = 0; e = kk; }
    else if (kk < 16) { g = 1; e = kk - 8; }
    else if (kk < 24) { g = 0; e = kk - 8; }
    else              { g = 1; e = kk - 16; }
    int lane = g * 16 + m;
    long d = (((long)mtile * KB + kb) * 32 + lane) * 16 + e;
    Ap[d] = (__bf16)X[i];
  }
}

// B [K x N] row-major -> Bp fragments: frag (ntile, kb) = 32 lanes x v16bf.
// Wave32 16-bit B layout: lane = g*16 + n ; g = (kk/8)%2 ; element e = kk%8 + 8*(kk/16).
__global__ void pack_B_bf16(const float* __restrict__ W, __bf16* __restrict__ Bp,
                            long total, int K, int N) {
  long i = (long)blockIdx.x * blockDim.x + threadIdx.x;
  long s = (long)gridDim.x * blockDim.x;
  int KB = K >> 5;
  for (; i < total; i += s) {
    int k = (int)(i / N), n = (int)(i % N);
    int ntile = n >> 4, nn = n & 15;
    int kb = k >> 5, kk = k & 31;
    int g = (kk >> 3) & 1, e = (kk & 7) | ((kk >> 4) << 3);
    int lane = g * 16 + nn;
    long d = (((long)ntile * KB + kb) * 32 + lane) * 16 + e;
    Bp[d] = (__bf16)W[i];
  }
}

// ---------------- WMMA GEMM on packed fragments ----------------
// C[m,n] = rowscale[m] * sum_k A[m,k]*B[k,n]
// Grid: (M/16)/MT blocks; block = (N/16) waves; wave w owns N-tile w and MT M-tiles.
// A fragments for the block's MT tiles are staged through LDS (async on gfx1250).
template <int K, int N, int MT>
__global__ __launch_bounds__(32 * (N / 16)) void wmma_gemm_packed(
    const __bf16* __restrict__ Ap, const __bf16* __restrict__ Bp,
    const float* __restrict__ rowscale, float* __restrict__ C) {
  constexpr int KB = K / 32;
  constexpr int NW = N / 16;
  constexpr int CHUNKS = MT * 64;  // 16-byte chunks per K-step (MT frags * 1KB)

  const int tid  = threadIdx.x;
  const int lane = tid & 31;
  const int wv   = tid >> 5;  // this wave's N-tile
  const int m0   = blockIdx.x * MT;
  const int half = lane & 15;
  const int grp  = lane >> 4;
  const int col  = wv * 16 + half;

  __shared__ __align__(128) __bf16 ldsA[MT * 512];  // MT fragments

  const v16bf* __restrict__ Bf = ((const v16bf*)Bp) + (long)wv * KB * 32 + lane;

  v8f acc[MT] = {};

#pragma unroll
  for (int kb = 0; kb < KB; ++kb) {
    // ---- stage MT A fragments for this K-step into LDS ----
    for (int c = tid; c < CHUNKS; c += 32 * NW) {
      int t = c >> 6, w = c & 63;
      const char* g = (const char*)Ap + (((long)(m0 + t) * KB + kb) * 1024) + (long)w * 16;
      char* l = (char*)ldsA + (long)c * 16;
#if USE_ASYNC_LDS
      AS1 vi4* gp = (AS1 vi4*)((AS1 void*)(void*)const_cast<char*>(g));
      AS3 vi4* lp = (AS3 vi4*)((AS3 void*)(void*)l);
      __builtin_amdgcn_global_load_async_to_lds_b128(gp, lp, 0, 0);
#else
      *(uint4*)l = *(const uint4*)g;
#endif
    }
#if USE_ASYNC_LDS
    __builtin_amdgcn_s_wait_asynccnt(0);
#endif
    __syncthreads();

    v16bf b = Bf[(long)kb * 32];
#pragma unroll
    for (int t = 0; t < MT; ++t) {
      v16bf a = ((const v16bf*)ldsA)[t * 32 + lane];
      acc[t] = __builtin_amdgcn_wmma_f32_16x16x32_bf16(
          /*neg_a=*/false, a, /*neg_b=*/false, b,
          /*c_mod=*/(short)0, acc[t], /*reuse_a=*/false, /*reuse_b=*/false);
    }
    __syncthreads();  // before next K-step overwrites ldsA
  }

  // C/D layout: VGPR r -> M = grp*8 + r, N = lane&15
#pragma unroll
  for (int t = 0; t < MT; ++t) {
#pragma unroll
    for (int r = 0; r < 8; ++r) {
      int m = (m0 + t) * 16 + grp * 8 + r;
      C[(long)m * N + col] = acc[t][r] * rowscale[m];
    }
  }
}

// ---------------- edge scatter: agg[dst] += msg[src] (msg already norm_out-scaled) ----------------
__global__ void scatter_kernel(const float4* __restrict__ msg4, const int* __restrict__ src,
                               const int* __restrict__ dst, float* __restrict__ agg,
                               int E, int Fc /*F/4*/, int F) {
  long t = (long)blockIdx.x * blockDim.x + threadIdx.x;
  if (t >= (long)E * Fc) return;
  int e = (int)(t / Fc);
  int c = (int)(t % Fc);
  float4 v = msg4[(long)src[e] * Fc + c];
  float* o = agg + (long)dst[e] * F + c * 4;
  atomicAdd(o + 0, v.x);
  atomicAdd(o + 1, v.y);
  atomicAdd(o + 2, v.z);
  atomicAdd(o + 3, v.w);
}

// ---------------- epilogue: g = relu(agg * norm_in[row] + bias[f]) in place ----------------
__global__ void relu_bias_norm(float* __restrict__ g, const float* __restrict__ norm_in,
                               const float* __restrict__ bias, long n, int F) {
  long i = (long)blockIdx.x * blockDim.x + threadIdx.x;
  long s = (long)gridDim.x * blockDim.x;
  for (; i < n; i += s) {
    int row = (int)(i / F);
    int f   = (int)(i % F);
    float x = g[i] * norm_in[row] + bias[f];
    g[i] = x > 0.0f ? x : 0.0f;
  }
}

// ---------------- final FC: out[o] = sum_n g2[n,o]*fc_w[n] + fc_b ----------------
__global__ void init_out(float* __restrict__ out, const float* __restrict__ fc_b) {
  if (threadIdx.x < OUTF) out[threadIdx.x] = fc_b[0];
}

__global__ void fc_kernel(const float* __restrict__ g2, const float* __restrict__ fc_w,
                          float* __restrict__ out) {
  int o     = threadIdx.x & 31;
  int nper  = blockDim.x >> 5;
  int nid   = blockIdx.x * nper + (threadIdx.x >> 5);
  int strid = gridDim.x * nper;
  float acc = 0.0f;
  for (int n = nid; n < NN; n += strid) acc += g2[(long)n * OUTF + o] * fc_w[n];
  atomicAdd(&out[o], acc);
}

// ---------------- host ----------------

static inline int gridFor(long n, int block) { return (int)((n + block - 1) / block); }

extern "C" void kernel_launch(void* const* d_in, const int* in_sizes, int n_in,
                              void* d_out, int out_size, void* d_ws, size_t ws_size,
                              hipStream_t stream) {
  const float* x    = (const float*)d_in[0];
  const int*   src  = (const int*)d_in[1];
  const int*   dst  = (const int*)d_in[2];
  const float* W1   = (const float*)d_in[3];
  const float* b1   = (const float*)d_in[4];
  const float* W2   = (const float*)d_in[5];
  const float* b2   = (const float*)d_in[6];
  const float* fc_w = (const float*)d_in[7];
  const float* fc_b = (const float*)d_in[8];
  float* out = (float*)d_out;

  (void)in_sizes; (void)n_in; (void)out_size; (void)ws_size;

  char* ws = (char*)d_ws;
  size_t off = 0;
  auto take = [&](size_t bytes) -> void* {
    void* p = ws + off;
    off += (bytes + 255) & ~(size_t)255;
    return p;
  };

  float*  norm_out = (float*)take((size_t)NN * 4);
  float*  norm_in  = (float*)take((size_t)NN * 4);
  __bf16* ApX  = (__bf16*)take((size_t)NN * INF * 2);   // packed A fragments, layer 1
  __bf16* Bp1  = (__bf16*)take((size_t)INF * HID * 2);  // packed B fragments, layer 1
  __bf16* Bp2  = (__bf16*)take((size_t)HID * OUTF * 2); // packed B fragments, layer 2
  float*  xw1  = (float*)take((size_t)NN * HID * 4);
  float*  g1   = (float*)take((size_t)NN * HID * 4);
  __bf16* Ap1  = (__bf16*)take((size_t)NN * HID * 2);   // packed A fragments, layer 2
  float*  xw2  = (float*)take((size_t)NN * OUTF * 4);
  float*  g2   = (float*)take((size_t)NN * OUTF * 4);

  const int B = 256;
  constexpr int MT = 5;  // M-tiles per wave; 3125 = 5 * 625

  // degrees -> norms
  zero_f32<<<gridFor(NN, B), B, 0, stream>>>(norm_out, NN);
  zero_f32<<<gridFor(NN, B), B, 0, stream>>>(norm_in, NN);
  degree_kernel<<<gridFor(NE, B), B, 0, stream>>>(src, dst, norm_out, norm_in, NE);
  norm_kernel<<<gridFor(NN, B), B, 0, stream>>>(norm_out, NN);
  norm_kernel<<<gridFor(NN, B), B, 0, stream>>>(norm_in, NN);

  // fragment packing (fused bf16 conversion)
  pack_A_bf16<<<gridFor((long)NN * INF, B), B, 0, stream>>>(x, ApX, (long)NN * INF, INF);
  pack_B_bf16<<<gridFor((long)INF * HID, B), B, 0, stream>>>(W1, Bp1, (long)INF * HID, INF, HID);
  pack_B_bf16<<<gridFor((long)HID * OUTF, B), B, 0, stream>>>(W2, Bp2, (long)HID * OUTF, HID, OUTF);

  // ---- layer 1: xw1 = (X @ W1) * norm_out[:,None] ----
  wmma_gemm_packed<INF, HID, MT>
      <<<(NN / 16) / MT, 32 * (HID / 16), 0, stream>>>(ApX, Bp1, norm_out, xw1);
  zero_f32<<<gridFor((long)NN * HID, B), B, 0, stream>>>(g1, (long)NN * HID);
  scatter_kernel<<<gridFor((long)NE * (HID / 4), B), B, 0, stream>>>(
      (const float4*)xw1, src, dst, g1, NE, HID / 4, HID);
  relu_bias_norm<<<gridFor((long)NN * HID, B), B, 0, stream>>>(g1, norm_in, b1,
                                                              (long)NN * HID, HID);

  // ---- layer 2: xw2 = (g1 @ W2) * norm_out[:,None] ----
  pack_A_bf16<<<gridFor((long)NN * HID, B), B, 0, stream>>>(g1, Ap1, (long)NN * HID, HID);
  wmma_gemm_packed<HID, OUTF, MT>
      <<<(NN / 16) / MT, 32 * (OUTF / 16), 0, stream>>>(Ap1, Bp2, norm_out, xw2);
  zero_f32<<<gridFor((long)NN * OUTF, B), B, 0, stream>>>(g2, (long)NN * OUTF);
  scatter_kernel<<<gridFor((long)NE * (OUTF / 4), B), B, 0, stream>>>(
      (const float4*)xw2, src, dst, g2, NE, OUTF / 4, OUTF);
  relu_bias_norm<<<gridFor((long)NN * OUTF, B), B, 0, stream>>>(g2, norm_in, b2,
                                                               (long)NN * OUTF, OUTF);

  // ---- final FC over nodes ----
  init_out<<<1, 32, 0, stream>>>(out, fc_b);
  fc_kernel<<<512, 256, 0, stream>>>(g2, fc_w, out);
}
